// MultiheadAttention_43061342109723
// MI455X (gfx1250) — matrix operand order repaired and checked by reading, not compiled, not run on hardware
//
#include <hip/hip_runtime.h>
#include <hip/hip_bf16.h>
#include <math.h>

// Problem constants (B, S, H, NH) = (2, 2048, 1024, 16), HD = 64
#define BB  2
#define SS  2048
#define HHH 1024
#define NHH 16
#define HDD 64
#define MM  (BB * SS)   // 4096 rows of the "token" matrix

typedef __attribute__((ext_vector_type(16))) __bf16       v16bf;
typedef __attribute__((ext_vector_type(8)))  float        v8f;
typedef __attribute__((ext_vector_type(4)))  unsigned int v4u;

union Frag { v16bf bf; v4u q[2]; unsigned int u[8]; };

__device__ inline unsigned f2bf_u(float f) {
  unsigned u = __float_as_uint(f);
  return (u + 0x7fffu + ((u >> 16) & 1u)) >> 16;   // RNE fp32 -> bf16
}
__device__ inline unsigned pk2(float lo, float hi) {
  return (f2bf_u(lo) & 0xffffu) | (f2bf_u(hi) << 16);
}
__device__ inline v8f wmma_bf16(v16bf a, v16bf b, v8f c) {
  // D = A(16x32 bf16) * B(32x16 bf16) + C(16x16 f32)
  return __builtin_amdgcn_wmma_f32_16x16x32_bf16(false, a, false, b, (short)0, c, false, false);
}
// xor-16 lane swap within wave32 (ds_swizzle_b32, group-of-32: xor=0x10, and=0x1f)
__device__ inline int   swz16(int v)    { return __builtin_amdgcn_ds_swizzle(v, 0x401f); }
__device__ inline float xmax16(float v) { return fmaxf(v, __int_as_float(swz16(__float_as_int(v)))); }
__device__ inline float xadd16(float v) { return v + __int_as_float(swz16(__float_as_int(v))); }

// ---------------------------------------------------------------- conversions
__global__ __launch_bounds__(256) void k_f32_to_bf16(const float* __restrict__ in,
                                                     unsigned short* __restrict__ out,
                                                     int n) {
  int i = blockIdx.x * 256 + threadIdx.x;
  if (i < n) out[i] = (unsigned short)f2bf_u(in[i]);
}

// W is K x N row-major fp32; Wt is N x K row-major bf16 (i.e. transposed)
__global__ __launch_bounds__(256) void k_transpose_bf16(const float* __restrict__ W,
                                                        unsigned short* __restrict__ Wt,
                                                        int Kdim, int N) {
  int i = blockIdx.x * 256 + threadIdx.x;
  if (i >= Kdim * N) return;
  int k = i / N, n = i % N;
  Wt[(size_t)n * Kdim + k] = (unsigned short)f2bf_u(W[i]);
}

// ---------------------------------------------------------------- bf16 GEMM
// One wave computes a 32(M) x 64(N) tile: each B fragment feeds 2 WMMAs
// (12 b128 loads per 8 WMMAs).
// C = A(M x K, bf16 row-major) * Wt(N x K, bf16 n-major)^T + bias
// mode 0: store bf16 row-major (ld = 1024)   [Q, K projections]
// mode 1: store bf16 transposed-per-head Vt[b][h][d][s]   [V projection]
// mode 2: store fp32 row-major (ld = 1024)   [final output projection]
__global__ __launch_bounds__(128) void k_gemm_bf16(const unsigned short* __restrict__ A,
                                                   const unsigned short* __restrict__ Wt,
                                                   const float* __restrict__ bias,
                                                   void* __restrict__ outp,
                                                   int K, int mode) {
  const int lane  = threadIdx.x;          // 0..31 (wave32)
  const int lq    = lane & 15;
  const int hi    = lane >> 4;            // 0: lanes 0-15, 1: lanes 16-31
  const int mbase = blockIdx.x * 32;
  const int nbase = (blockIdx.y * 4 + threadIdx.y) * 64;

  v8f acc[2][4] = {};
  const unsigned short* arow0 = A + (size_t)(mbase + lq) * K;
  const unsigned short* arow1 = arow0 + (size_t)16 * K;
  const unsigned short* wcol  = Wt + (size_t)(nbase + lq) * K;

  for (int kk = 0; kk < K; kk += 32) {
    __builtin_prefetch(arow0 + kk + 128, 0, 1);
    __builtin_prefetch(arow1 + kk + 128, 0, 1);
    Frag a0, a1;
    const int ao = kk + (hi ? 8 : 0);        // A-frag: j0-7 -> k=ao.., j8-15 -> k=ao+16..
    a0.q[0] = *(const v4u*)(arow0 + ao);
    a0.q[1] = *(const v4u*)(arow0 + ao + 16);
    a1.q[0] = *(const v4u*)(arow1 + ao);
    a1.q[1] = *(const v4u*)(arow1 + ao + 16);

    const int bo = kk + (hi ? 8 : 0);        // B-frag: j0-7 -> k=bo.., j8-15 -> k=bo+16..
#pragma unroll
    for (int t = 0; t < 4; ++t) {
      Frag b;
      const unsigned short* w = wcol + (size_t)(t * 16) * K + bo;
      b.q[0] = *(const v4u*)(w);
      b.q[1] = *(const v4u*)(w + 16);
      acc[0][t] = wmma_bf16(a0.bf, b.bf, acc[0][t]);
      acc[1][t] = wmma_bf16(a1.bf, b.bf, acc[1][t]);
    }
  }

#pragma unroll
  for (int r = 0; r < 2; ++r) {
#pragma unroll
    for (int t = 0; t < 4; ++t) {
      const int n  = nbase + t * 16 + lq;    // output column held by this lane
      const float bv = bias[n];
#pragma unroll
      for (int i = 0; i < 8; ++i) {
        const int m   = mbase + r * 16 + hi * 8 + i;  // C layout: VGPR i -> row i (+8 hi)
        const float v = acc[r][t][i] + bv;
        if (mode == 0) {
          ((unsigned short*)outp)[(size_t)m * HHH + n] = (unsigned short)f2bf_u(v);
        } else if (mode == 1) {
          const int h = n >> 6, d = n & 63;
          const int b = m >> 11, s = m & (SS - 1);
          ((unsigned short*)outp)[(((size_t)(b * NHH + h) * HDD + d) * SS) + s] =
              (unsigned short)f2bf_u(v);
        } else {
          ((float*)outp)[(size_t)m * HHH + n] = v;
        }
      }
    }
  }
}

// ---------------------------------------------------------------- attention
// One wave owns one (b, h, 32-query tile): K/V fragments are reused by both
// 16-query sub-tiles (16 b128 loads per 16 WMMAs). Flash-style online softmax.
// Computes S^T = K * Q^T (rows = keys, cols = queries) so per-query softmax
// stats live in-lane (+ one xor16 swizzle), and O^T = V^T * P^T.
__global__ __launch_bounds__(128) void k_attention(const unsigned short* __restrict__ Qb,
                                                   const unsigned short* __restrict__ Kb,
                                                   const unsigned short* __restrict__ Vt,
                                                   unsigned short* __restrict__ Ab) {
  const int lane  = threadIdx.x;
  const int lq    = lane & 15;
  const int hi    = lane >> 4;
  const int b     = blockIdx.z;
  const int h     = blockIdx.y;
  const int qbase = (blockIdx.x * 4 + threadIdx.y) * 32;
  const float cs  = 0.125f * 1.44269504088896f;   // 1/sqrt(64) * log2(e)

  // Q^T B-fragments (loaded once): qt sub-tile, d-split; col q = qbase+qt*16+lq
  Frag qb[2][2];
  {
    const int off = hi ? 8 : 0;
#pragma unroll
    for (int qt = 0; qt < 2; ++qt) {
      const unsigned short* qrow =
          Qb + (size_t)(b * SS + qbase + qt * 16 + lq) * HHH + h * HDD;
      qb[qt][0].q[0] = *(const v4u*)(qrow + off);
      qb[qt][0].q[1] = *(const v4u*)(qrow + off + 16);
      qb[qt][1].q[0] = *(const v4u*)(qrow + 32 + off);
      qb[qt][1].q[1] = *(const v4u*)(qrow + 32 + off + 16);
    }
  }

  float m[2] = {-3.0e38f, -3.0e38f}, l[2] = {0.0f, 0.0f};
  v8f acc[2][4] = {};   // O^T per q sub-tile: 4 d-tiles x (16 x 16q)
  const unsigned short* kbasep = Kb + (size_t)(b * SS) * HHH + h * HDD;
  const unsigned short* vbasep = Vt + (size_t)((b * NHH + h) * HDD) * SS;

  for (int kt = 0; kt < SS; kt += 32) {
    const int o = hi ? 8 : 0;
    const unsigned short* kr0 = kbasep + (size_t)(kt + lq) * HHH;
    const unsigned short* kr1 = kr0 + (size_t)16 * HHH;
    __builtin_prefetch(kr0 + (size_t)32 * HHH, 0, 1);

    Frag ka[2][2];   // [key-row-tile][d-split]
    ka[0][0].q[0] = *(const v4u*)(kr0 + o);       ka[0][0].q[1] = *(const v4u*)(kr0 + o + 16);
    ka[0][1].q[0] = *(const v4u*)(kr0 + 32 + o);  ka[0][1].q[1] = *(const v4u*)(kr0 + 32 + o + 16);
    ka[1][0].q[0] = *(const v4u*)(kr1 + o);       ka[1][0].q[1] = *(const v4u*)(kr1 + o + 16);
    ka[1][1].q[0] = *(const v4u*)(kr1 + 32 + o);  ka[1][1].q[1] = *(const v4u*)(kr1 + 32 + o + 16);

    v8f st[2][2];    // S^T tiles: [key-row-tile][q sub-tile]
#pragma unroll
    for (int r = 0; r < 2; ++r) {
#pragma unroll
      for (int qt = 0; qt < 2; ++qt) {
        v8f z = {};
        z = wmma_bf16(ka[r][0].bf, qb[qt][0].bf, z);
        st[r][qt] = wmma_bf16(ka[r][1].bf, qb[qt][1].bf, z);
      }
    }

    Frag pt[2];
    float scale_old[2];
#pragma unroll
    for (int qt = 0; qt < 2; ++qt) {
      // Online softmax update for this lane's query column (sub-tile qt)
      float mb = st[0][qt][0];
#pragma unroll
      for (int i = 1; i < 8; ++i) mb = fmaxf(mb, st[0][qt][i]);
#pragma unroll
      for (int i = 0; i < 8; ++i) mb = fmaxf(mb, st[1][qt][i]);
      mb = xmax16(mb);
      const float mn = fmaxf(m[qt], mb);
      scale_old[qt] = exp2f((m[qt] - mn) * cs);

      float p0[8], p1[8], ls = 0.0f;
#pragma unroll
      for (int i = 0; i < 8; ++i) { p0[i] = exp2f((st[0][qt][i] - mn) * cs); ls += p0[i]; }
#pragma unroll
      for (int i = 0; i < 8; ++i) { p1[i] = exp2f((st[1][qt][i] - mn) * cs); ls += p1[i]; }
      ls = xadd16(ls);
      l[qt] = l[qt] * scale_old[qt] + ls;
      m[qt] = mn;

      // Build P^T B-fragment: low lanes own keys kt+0..7 (p0), need kt+8..15
      // from partner lane (+16); high lanes own kt+24..31 (p1), need kt+16..23.
#pragma unroll
      for (int j = 0; j < 4; ++j) {
        const unsigned lo  = pk2(p0[2 * j], p0[2 * j + 1]);
        const unsigned hi2 = pk2(p1[2 * j], p1[2 * j + 1]);
        const unsigned rcv = (unsigned)swz16((int)(hi ? lo : hi2));
        pt[qt].u[j]     = hi ? rcv : lo;
        pt[qt].u[4 + j] = hi ? hi2 : rcv;
      }

      // Rescale accumulators for this q sub-tile
#pragma unroll
      for (int t = 0; t < 4; ++t)
#pragma unroll
        for (int i = 0; i < 8; ++i) acc[qt][t][i] *= scale_old[qt];
    }

    // O^T += V^T-tile * P^T ; each V fragment feeds both q sub-tiles
#pragma unroll
    for (int t = 0; t < 4; ++t) {
      Frag va;
      const unsigned short* vr = vbasep + (size_t)(t * 16 + lq) * SS + kt;
      va.q[0] = *(const v4u*)(vr + 16 * hi);
      va.q[1] = *(const v4u*)(vr + 16 * hi + 8);
      acc[0][t] = wmma_bf16(va.bf, pt[0].bf, acc[0][t]);
      acc[1][t] = wmma_bf16(va.bf, pt[1].bf, acc[1][t]);
    }
  }

#pragma unroll
  for (int qt = 0; qt < 2; ++qt) {
    const float inv = 1.0f / l[qt];
    const int q = qbase + qt * 16 + lq;
#pragma unroll
    for (int t = 0; t < 4; ++t) {
#pragma unroll
      for (int i = 0; i < 8; ++i) {
        const int d = t * 16 + hi * 8 + i;
        Ab[(size_t)(b * SS + q) * HHH + h * HDD + d] =
            (unsigned short)f2bf_u(acc[qt][t][i] * inv);
      }
    }
  }
}

// ---------------------------------------------------------------- launcher
extern "C" void kernel_launch(void* const* d_in, const int* in_sizes, int n_in,
                              void* d_out, int out_size, void* d_ws, size_t ws_size,
                              hipStream_t stream) {
  const float* x   = (const float*)d_in[0];  // (B,S,H)
  const float* Wq  = (const float*)d_in[1];  // (H,H)
  const float* bq  = (const float*)d_in[2];  // (H)
  const float* Wkv = (const float*)d_in[3];  // (H,2H)
  const float* bkv = (const float*)d_in[4];  // (2H)
  const float* Wo  = (const float*)d_in[5];  // (H,H)
  const float* bo  = (const float*)d_in[6];  // (H)

  char* ws = (char*)d_ws;
  unsigned short* Xb   = (unsigned short*)(ws + (size_t)0);            // 8 MiB
  unsigned short* Wqt  = (unsigned short*)(ws + ((size_t)8  << 20));   // 2 MiB
  unsigned short* Wkvt = (unsigned short*)(ws + ((size_t)10 << 20));   // 4 MiB
  unsigned short* Wot  = (unsigned short*)(ws + ((size_t)14 << 20));   // 2 MiB
  unsigned short* Qb   = (unsigned short*)(ws + ((size_t)16 << 20));   // 8 MiB
  unsigned short* Kb   = (unsigned short*)(ws + ((size_t)24 << 20));   // 8 MiB
  unsigned short* Vt   = (unsigned short*)(ws + ((size_t)32 << 20));   // 8 MiB
  unsigned short* Ab   = (unsigned short*)(ws + ((size_t)40 << 20));   // 8 MiB

  // 0) bf16 conversion / weight transposition
  k_f32_to_bf16<<<(MM * HHH) / 256, 256, 0, stream>>>(x, Xb, MM * HHH);
  k_transpose_bf16<<<(HHH * HHH) / 256, 256, 0, stream>>>(Wq, Wqt, HHH, HHH);
  k_transpose_bf16<<<(HHH * 2 * HHH) / 256, 256, 0, stream>>>(Wkv, Wkvt, HHH, 2 * HHH);
  k_transpose_bf16<<<(HHH * HHH) / 256, 256, 0, stream>>>(Wo, Wot, HHH, HHH);

  // 1) projections: each wave does a 32x64 tile; grid = (M/32, N/(4*64))
  dim3 gblk(32, 4, 1);
  dim3 ggrd(MM / 32, HHH / 256, 1);
  k_gemm_bf16<<<ggrd, gblk, 0, stream>>>(Xb, Wqt, bq, (void*)Qb, HHH, 0);
  k_gemm_bf16<<<ggrd, gblk, 0, stream>>>(Xb, Wkvt, bkv, (void*)Kb, HHH, 0);
  k_gemm_bf16<<<ggrd, gblk, 0, stream>>>(Xb, Wkvt + (size_t)HHH * HHH, bkv + HHH,
                                         (void*)Vt, HHH, 1);

  // 2) attention: one wave per (b, h, 32-query tile)
  dim3 ablk(32, 4, 1);
  dim3 agrd(SS / 128, NHH, BB);
  k_attention<<<agrd, ablk, 0, stream>>>(Qb, Kb, Vt, Ab);

  // 3) output projection -> fp32
  k_gemm_bf16<<<ggrd, gblk, 0, stream>>>(Ab, Wot, bo, d_out, HHH, 2);
  (void)in_sizes; (void)n_in; (void)out_size; (void)ws_size;
}
